// CloudCrop_17910013624408
// MI455X (gfx1250) — compile-verified, matrix operand order checked by reading, not run on hardware
//
#include <hip/hip_runtime.h>
#include <hip/hip_fp16.h>

typedef __attribute__((ext_vector_type(16))) _Float16 v16h;
typedef __attribute__((ext_vector_type(8)))  float    v8f;

#define B_DIM   2
#define S_DIM   1024
#define N_PTS   10000
#define C_FEAT  128
#define K1      160          // 131 padded to 5*32
#define NS      64           // NSAMPLE
#define RAD2    (0.05f*0.05f)
#define HMIN_   (-0.02f)
#define HMAX_   (0.02f)
#define BN_INV  0.9999950000374997f   // 1/sqrt(1+1e-5)

// padded LDS leading dims (halves; multiples of 8 keep 16B alignment, +8 breaks
// the 512B-stride bank pattern)
#define GLD  168   // g   : 64 x 160  -> ld 168
#define XLD  264   // x   : 64 x 256  -> ld 264
#define QLD  72    // q   : 64 x 64   -> ld 72
#define ALD  72    // attT: 64 x 64
#define VLD  72    // vT  : 256 x 64
#define YLD  264   // y   : 64 x 256
#define SLD  66    // sAtt (f32): 64 x 64 -> ld 66 floats

// LDS layout (bytes)
#define OFF_Q     0
#define OFF_S     9216                 // 64*72*2 = 9216 for q
#define OFF_AT    26112                // 9216 + 64*66*4(=16896)
#define OFF_G     0                    // g overlaps q/s/attT region (dead before q)
#define OFF_X     35328                // region0 = max(21504, 35328)
#define OFF_VT    69120                // 35328 + 64*264*2(=33792)
#define OFF_Y     105984               // + 256*72*2(=36864)
#define OFF_IDX   139776               // + 33792
#define OFF_REL   140032
#define OFF_SCAN  140800
#define OFF_POOL  140832
#define SMEM_BYTES 141888

// d_ws layout (halves): [Wmlp 256*160 | Wqk 64*256 | Wv 256*256 | Wt 256*256 | upT]
#define W_TOT_H   (256*K1 + 64*256 + 256*256 + 256*256)     // 188416 halves
#define UPT_H     (B_DIM * N_PTS * C_FEAT)                  // 2,560,000 halves
#define WS_NEEDED ((size_t)(W_TOT_H + UPT_H) * 2)

union V16U { uint4 u[2]; v16h v; };

// Load a 16x32 f16 tile in the CDNA5 WMMA 16-bit A/B register layout.
// lane 0-15 : row = lane,    K = 0..7 (vgpr0-3, pairs) and 16..23 (vgpr4-7)
// lane16-31 : row = lane-16, K = 8..15 and 24..31
__device__ __forceinline__ v16h ld_tile(const _Float16* base, int ld) {
  const int lane = threadIdx.x & 31;
  const _Float16* p = base + (lane & 15) * ld + ((lane >> 4) << 3);
  V16U t;
  t.u[0] = *(const uint4*)(p);
  t.u[1] = *(const uint4*)(p + 16);
  return t.v;
}

__device__ __forceinline__ v8f wmma16(v16h a, v16h b, v8f c) {
  return __builtin_amdgcn_wmma_f32_16x16x32_f16(false, a, false, b, (short)0, c,
                                                false, false);
}

// --------- weight f32 -> f16 conversion; mlp columns permuted to -----------
// --------- [feat(128) | xyz(3) | zero pad] to match g's column order -------
__global__ void convert_weights(const float* __restrict__ mlp_w,
                                const float* __restrict__ qk_w,
                                const float* __restrict__ v_w,
                                const float* __restrict__ t_w,
                                _Float16* __restrict__ ws) {
  const int NM = 256 * K1;        // 40960  (padded, permuted mlp)
  const int NQ = 64 * 256;        // 16384
  const int NV = 256 * 256;
  const int NT = 256 * 256;
  int i = blockIdx.x * blockDim.x + threadIdx.x;
  if (i < NM) {
    int o = i / K1, c = i - o * K1;
    float v;
    if (c < 128)        v = mlp_w[o * 131 + 3 + c];   // feature weights
    else if (c < 131)   v = mlp_w[o * 131 + (c - 128)]; // xyz weights
    else                v = 0.f;
    ws[i] = (_Float16)v;
  } else if (i < NM + NQ) {
    ws[i] = (_Float16)qk_w[i - NM];
  } else if (i < NM + NQ + NV) {
    ws[i] = (_Float16)v_w[i - NM - NQ];
  } else if (i < NM + NQ + NV + NT) {
    ws[i] = (_Float16)t_w[i - NM - NQ - NV];
  }
}

// --------- up_feature (B,C,N) f32  ->  upT (B,N,C) f16 (coalesced reads) ----
__global__ void transpose_features(const float* __restrict__ up,
                                   _Float16* __restrict__ upT) {
  int i = blockIdx.x * blockDim.x + threadIdx.x;   // linear over (b,c,n)
  if (i >= B_DIM * C_FEAT * N_PTS) return;
  int b = i / (C_FEAT * N_PTS);
  int rem = i - b * (C_FEAT * N_PTS);
  int c = rem / N_PTS;
  int n = rem - c * N_PTS;
  upT[((size_t)b * N_PTS + n) * C_FEAT + c] = (_Float16)up[i];
}

// ---------------- main kernel: one workgroup (8 waves) per seed --------------
__global__ void __launch_bounds__(256)
cloudcrop_kernel(const float* __restrict__ seed_xyz,
                 const float* __restrict__ pointcloud,
                 const float* __restrict__ vp_rot,
                 const float* __restrict__ up_feature,
                 const float* __restrict__ mlp_b,
                 const float* __restrict__ bn1_g, const float* __restrict__ bn1_b,
                 const float* __restrict__ v_b,  const float* __restrict__ t_b,
                 const float* __restrict__ bn2_g, const float* __restrict__ bn2_b,
                 const _Float16* __restrict__ wgt,
                 const _Float16* __restrict__ upT, int haveT,
                 float* __restrict__ out) {
  extern __shared__ char smem[];
  _Float16* gB   = (_Float16*)(smem + OFF_G);
  _Float16* qB   = (_Float16*)(smem + OFF_Q);
  float*    sAtt = (float*)   (smem + OFF_S);
  _Float16* attT = (_Float16*)(smem + OFF_AT);
  _Float16* xB   = (_Float16*)(smem + OFF_X);
  _Float16* vT   = (_Float16*)(smem + OFF_VT);
  _Float16* yB   = (_Float16*)(smem + OFF_Y);
  int*      idxbuf  = (int*)     (smem + OFF_IDX);
  float*    relbuf  = (float*)   (smem + OFF_REL);
  int*      scanbuf = (int*)     (smem + OFF_SCAN);
  unsigned* pooled  = (unsigned*)(smem + OFF_POOL);

  const int tid  = threadIdx.x;
  const int lane = tid & 31;
  const int wv   = tid >> 5;
  const int bs   = blockIdx.x;
  const int b    = bs >> 10;
  const int s    = bs & 1023;

  pooled[tid] = 0u;   // final values are >= 0; uint-max == float-max

  const float sx = seed_xyz[(b * S_DIM + s) * 3 + 0];
  const float sy = seed_xyz[(b * S_DIM + s) * 3 + 1];
  const float sz = seed_xyz[(b * S_DIM + s) * 3 + 2];
  float R[9];
  #pragma unroll
  for (int k = 0; k < 9; k++) R[k] = vp_rot[((size_t)(b * S_DIM + s)) * 9 + k];

  // ---- stage 1: cylinder query, ordered compaction of first 64 indices ----
  int cnt = 0;
  for (int base = 0; base < N_PTS && cnt < NS; base += 256) {
    const int i = base + tid;
    const bool valid = (i < N_PTS);
    float px = 0.f, py = 0.f, pz = 0.f;
    if (valid) {
      const float* p = pointcloud + ((size_t)b * N_PTS + i) * 3;
      px = p[0]; py = p[1]; pz = p[2];
    }
    const float dx = px - sx, dy = py - sy, dz = pz - sz;
    const float r0 = dx * R[0] + dy * R[3] + dz * R[6];
    const float r1 = dx * R[1] + dy * R[4] + dz * R[7];
    const float r2 = dx * R[2] + dy * R[5] + dz * R[8];
    const bool m = valid && (r1 * r1 + r2 * r2 < RAD2) && (r0 > HMIN_) && (r0 < HMAX_);

    const unsigned bal = __builtin_amdgcn_ballot_w32(m);
    if (lane == 0) scanbuf[wv] = __popc(bal);
    __syncthreads();
    int pre = 0, tot = 0;
    #pragma unroll
    for (int k = 0; k < 8; k++) { int c = scanbuf[k]; tot += c; if (k < wv) pre += c; }
    const int pos = cnt + pre + __popc(bal & ((1u << lane) - 1u));
    if (m && pos < NS) {
      idxbuf[pos] = i;
      relbuf[pos * 3 + 0] = r0;
      relbuf[pos * 3 + 1] = r1;
      relbuf[pos * 3 + 2] = r2;
    }
    cnt = min(NS, cnt + tot);
    __syncthreads();
  }
  if (cnt == 0) {                 // no hits: reference degrades to index 0
    if (tid == 0) {
      const float* p = pointcloud + (size_t)b * N_PTS * 3;
      const float dx = p[0] - sx, dy = p[1] - sy, dz = p[2] - sz;
      relbuf[0] = dx * R[0] + dy * R[3] + dz * R[6];
      relbuf[1] = dx * R[1] + dy * R[4] + dz * R[7];
      relbuf[2] = dx * R[2] + dy * R[5] + dz * R[8];
      idxbuf[0] = 0;
    }
    __syncthreads();
    cnt = 1;
  }
  for (int k = cnt + tid; k < NS; k += 256) {   // pad with first sample
    idxbuf[k] = idxbuf[0];
    relbuf[k * 3 + 0] = relbuf[0];
    relbuf[k * 3 + 1] = relbuf[1];
    relbuf[k * 3 + 2] = relbuf[2];
  }
  __syncthreads();

  // ---- stage 2: gather -> g (64 x 160 f16), cols: [feat | xyz | zero pad] ----
  if (haveT) {
    // point-major f16 features: 256B contiguous per sample, b128 transfers
    const _Float16* upTb = upT + (size_t)b * N_PTS * C_FEAT;
    for (int e = tid; e < NS * 16; e += 256) {
      const int k = e >> 4, c8 = e & 15;
      const uint4 d = *(const uint4*)(upTb + (size_t)idxbuf[k] * C_FEAT + c8 * 8);
      *(uint4*)(gB + k * GLD + c8 * 8) = d;
    }
  } else {
    // fallback: channel-major f32 gather
    const float* upb = up_feature + (size_t)b * C_FEAT * N_PTS;
    for (int e = tid; e < NS * C_FEAT; e += 256) {
      const int k = e >> 7, c = e & 127;
      gB[k * GLD + c] = (_Float16)upb[(size_t)c * N_PTS + idxbuf[k]];
    }
  }
  if (tid < NS) {
    const int k = tid;
    gB[k * GLD + 128] = (_Float16)relbuf[k * 3 + 0];
    gB[k * GLD + 129] = (_Float16)relbuf[k * 3 + 1];
    gB[k * GLD + 130] = (_Float16)relbuf[k * 3 + 2];
    for (int c = 131; c < K1; c++) gB[k * GLD + c] = (_Float16)0.f;
  }
  __syncthreads();

  const _Float16* Wmlp = wgt;
  const _Float16* Wqk  = Wmlp + 256 * K1;
  const _Float16* Wv   = Wqk + 64 * 256;
  const _Float16* Wt   = Wv + 256 * 256;

  const int n16 = lane & 15;
  const int m8  = (lane >> 4) << 3;

  // ---- stage 3: x = relu(bn1(g @ Wmlp^T + b))   (64x256, K=160) ----
  for (int t = wv; t < 64; t += 8) {
    const int mt = t >> 4, nt = t & 15;
    v8f acc = {};
    #pragma unroll
    for (int ks = 0; ks < 5; ks++)
      acc = wmma16(ld_tile(gB + mt * 16 * GLD + ks * 32, GLD),
                   ld_tile(Wmlp + nt * 16 * K1 + ks * 32, K1), acc);
    const int n0 = nt * 16 + n16, m0 = mt * 16 + m8;
    const float bb = mlp_b[n0], g1 = bn1_g[n0], b1 = bn1_b[n0];
    #pragma unroll
    for (int r = 0; r < 8; r++) {
      float h = fmaf((acc[r] + bb) * BN_INV, g1, b1);
      xB[(m0 + r) * XLD + n0] = (_Float16)fmaxf(h, 0.f);
    }
  }
  __syncthreads();

  // ---- stage 4: q = x @ Wqk^T   (64x64, K=256) ----
  for (int t = wv; t < 16; t += 8) {
    const int mt = t >> 2, nt = t & 3;
    v8f acc = {};
    #pragma unroll
    for (int ks = 0; ks < 8; ks++)
      acc = wmma16(ld_tile(xB + mt * 16 * XLD + ks * 32, XLD),
                   ld_tile(Wqk + nt * 16 * 256 + ks * 32, 256), acc);
    const int n0 = nt * 16 + n16, m0 = mt * 16 + m8;
    #pragma unroll
    for (int r = 0; r < 8; r++) qB[(m0 + r) * QLD + n0] = (_Float16)acc[r];
  }
  __syncthreads();

  // ---- stage 5: s = q @ q^T   (64x64, K=64) ----
  for (int t = wv; t < 16; t += 8) {
    const int mt = t >> 2, nt = t & 3;
    v8f acc = {};
    #pragma unroll
    for (int ks = 0; ks < 2; ks++)
      acc = wmma16(ld_tile(qB + mt * 16 * QLD + ks * 32, QLD),
                   ld_tile(qB + nt * 16 * QLD + ks * 32, QLD), acc);
    const int n0 = nt * 16 + n16, m0 = mt * 16 + m8;
    #pragma unroll
    for (int r = 0; r < 8; r++) sAtt[(m0 + r) * SLD + n0] = acc[r];
  }
  __syncthreads();

  // ---- stage 6: row softmax, then column renorm, write attT (f16) ----
  if (tid < NS) {
    float* row = sAtt + tid * SLD;
    float mx = -3.4e38f;
    for (int j = 0; j < NS; j++) mx = fmaxf(mx, row[j]);
    float sum = 0.f;
    for (int j = 0; j < NS; j++) { float e = __expf(row[j] - mx); row[j] = e; sum += e; }
    const float rs = 1.f / sum;
    for (int j = 0; j < NS; j++) row[j] *= rs;
  }
  __syncthreads();
  if (tid < NS) {
    const int j = tid;
    float cs = 0.f;
    for (int i = 0; i < NS; i++) cs += sAtt[i * SLD + j];
    const float rn = 1.f / (1e-9f + cs);
    for (int i = 0; i < NS; i++) attT[j * ALD + i] = (_Float16)(sAtt[i * SLD + j] * rn);
  }
  __syncthreads();

  // ---- stage 7: v = x @ Wv^T + v_b, stored transposed (256x64) ----
  for (int t = wv; t < 64; t += 8) {
    const int mt = t >> 4, nt = t & 15;
    v8f acc = {};
    #pragma unroll
    for (int ks = 0; ks < 8; ks++)
      acc = wmma16(ld_tile(xB + mt * 16 * XLD + ks * 32, XLD),
                   ld_tile(Wv + nt * 16 * 256 + ks * 32, 256), acc);
    const int n0 = nt * 16 + n16, m0 = mt * 16 + m8;
    const float vb = v_b[n0];
    #pragma unroll
    for (int r = 0; r < 8; r++) vT[n0 * VLD + m0 + r] = (_Float16)(acc[r] + vb);
  }
  __syncthreads();

  // ---- stage 8: x_r = attT @ v  (K=64);  y = x - x_r ----
  for (int t = wv; t < 64; t += 8) {
    const int mt = t >> 4, nt = t & 15;
    v8f acc = {};
    #pragma unroll
    for (int ks = 0; ks < 2; ks++)
      acc = wmma16(ld_tile(attT + mt * 16 * ALD + ks * 32, ALD),
                   ld_tile(vT + nt * 16 * VLD + ks * 32, VLD), acc);
    const int n0 = nt * 16 + n16, m0 = mt * 16 + m8;
    #pragma unroll
    for (int r = 0; r < 8; r++) {
      const float xv = (float)xB[(m0 + r) * XLD + n0];
      yB[(m0 + r) * YLD + n0] = (_Float16)(xv - acc[r]);
    }
  }
  __syncthreads();

  // ---- stage 9: t = y @ Wt^T + t_b; x2 = x + relu(bn2(t)); max-pool ----
  for (int t = wv; t < 64; t += 8) {
    const int mt = t >> 4, nt = t & 15;
    v8f acc = {};
    #pragma unroll
    for (int ks = 0; ks < 8; ks++)
      acc = wmma16(ld_tile(yB + mt * 16 * YLD + ks * 32, YLD),
                   ld_tile(Wt + nt * 16 * 256 + ks * 32, 256), acc);
    const int n0 = nt * 16 + n16, m0 = mt * 16 + m8;
    const float tb = t_b[n0], g2 = bn2_g[n0], b2 = bn2_b[n0];
    float lmax = 0.f;                       // x2 >= 0 always
    #pragma unroll
    for (int r = 0; r < 8; r++) {
      float tv = fmaf((acc[r] + tb) * BN_INV, g2, b2);
      tv = fmaxf(tv, 0.f);
      const float x2 = (float)xB[(m0 + r) * XLD + n0] + tv;
      lmax = fmaxf(lmax, x2);
    }
    atomicMax(&pooled[n0], __float_as_uint(lmax));   // ds_max_u32
  }
  __syncthreads();

  // ---- output: (B, 256, S) ----
  out[((size_t)b * 256 + tid) * S_DIM + s] = __uint_as_float(pooled[tid]);
}

extern "C" void kernel_launch(void* const* d_in, const int* in_sizes, int n_in,
                              void* d_out, int out_size, void* d_ws, size_t ws_size,
                              hipStream_t stream) {
  const float* seed_xyz   = (const float*)d_in[0];
  const float* pointcloud = (const float*)d_in[1];
  const float* vp_rot     = (const float*)d_in[2];
  const float* up_feature = (const float*)d_in[3];
  const float* mlp_w      = (const float*)d_in[4];
  const float* mlp_b      = (const float*)d_in[5];
  const float* bn1_g      = (const float*)d_in[6];
  const float* bn1_b      = (const float*)d_in[7];
  const float* qk_w       = (const float*)d_in[8];
  const float* v_w        = (const float*)d_in[9];
  const float* v_b        = (const float*)d_in[10];
  const float* t_w        = (const float*)d_in[11];
  const float* t_b        = (const float*)d_in[12];
  const float* bn2_g      = (const float*)d_in[13];
  const float* bn2_b      = (const float*)d_in[14];
  _Float16* wgt = (_Float16*)d_ws;
  _Float16* upT = wgt + W_TOT_H;
  float* out = (float*)d_out;

  const int haveT = (ws_size >= WS_NEEDED) ? 1 : 0;

  (void)hipFuncSetAttribute((const void*)cloudcrop_kernel,
                            hipFuncAttributeMaxDynamicSharedMemorySize, SMEM_BYTES);

  convert_weights<<<(W_TOT_H + 255) / 256, 256, 0, stream>>>(mlp_w, qk_w, v_w, t_w, wgt);
  if (haveT) {
    const int te = B_DIM * C_FEAT * N_PTS;
    transpose_features<<<(te + 255) / 256, 256, 0, stream>>>(up_feature, upT);
  }
  cloudcrop_kernel<<<B_DIM * S_DIM, 256, SMEM_BYTES, stream>>>(
      seed_xyz, pointcloud, vp_rot, up_feature, mlp_b, bn1_g, bn1_b,
      v_b, t_b, bn2_g, bn2_b, wgt, upT, haveT, out);
}
